// Attention_23991687315844
// MI455X (gfx1250) — matrix operand order repaired and checked by reading, not compiled, not run on hardware
//
#include <hip/hip_runtime.h>
#include <hip/hip_bf16.h>
#include <math.h>

// ---------------- configuration (matches reference) ----------------
#define B_      16
#define S_      2048
#define D_      512
#define QT      16            // q rows per workgroup (one WMMA M tile)
#define NWAVES  8
#define THREADS 256
#define KTILES  (S_ / 16)     // 128 key tiles
#define QK_CHUNKS (D_ / 32)   // 16 K-dim chunks for Q@K^T
#define PV_CHUNKS (S_ / 32)   // 64 K-dim chunks for P@V

typedef __bf16 bf16_t;
typedef bf16_t v16bf __attribute__((ext_vector_type(16)));
typedef float  v8f   __attribute__((ext_vector_type(8)));

// A/B operand for v_wmma_f32_16x16x32_bf16 from a row-major bf16 row in LDS.
// Lane group g = lane>>4. VGPRs 0-3 hold K = g*8+0..7, VGPRs 4-7 hold K = 16+g*8+0..7.
__device__ __forceinline__ v16bf ld16_bf16(const bf16_t* row, int g) {
    v16bf a;
    uint4* pa = reinterpret_cast<uint4*>(&a);
    pa[0] = *reinterpret_cast<const uint4*>(row + g * 8);        // 8 bf16
    pa[1] = *reinterpret_cast<const uint4*>(row + 16 + g * 8);   // 8 bf16
    return a;
}

// Same operand layout but sourced from an f32 row (global K rows, or LDS P rows),
// converting to bf16 on the fly.
__device__ __forceinline__ v16bf cvt16_f32(const float* row, int g) {
    const float4 f0 = *reinterpret_cast<const float4*>(row + g * 8);
    const float4 f1 = *reinterpret_cast<const float4*>(row + g * 8 + 4);
    const float4 f2 = *reinterpret_cast<const float4*>(row + 16 + g * 8);
    const float4 f3 = *reinterpret_cast<const float4*>(row + 16 + g * 8 + 4);
    v16bf r;
    r[0]  = (bf16_t)f0.x; r[1]  = (bf16_t)f0.y; r[2]  = (bf16_t)f0.z; r[3]  = (bf16_t)f0.w;
    r[4]  = (bf16_t)f1.x; r[5]  = (bf16_t)f1.y; r[6]  = (bf16_t)f1.z; r[7]  = (bf16_t)f1.w;
    r[8]  = (bf16_t)f2.x; r[9]  = (bf16_t)f2.y; r[10] = (bf16_t)f2.z; r[11] = (bf16_t)f2.w;
    r[12] = (bf16_t)f3.x; r[13] = (bf16_t)f3.y; r[14] = (bf16_t)f3.z; r[15] = (bf16_t)f3.w;
    return r;
}

extern __shared__ char smem[];

// LDS layout (dynamic):
//   sQ   : bf16 [16][512]            16 KB
//   sS   : f32  [16][2048]          128 KB   (scores -> exp values)
//   sVt  : bf16 [512][32] transposed 32 KB   (V chunk, K-major contiguous)
//   sRed : f32  [256]                 1 KB   (softmax reductions)
//   sInv : f32  [16]                          (1 / rowsum)
#define SMEM_BYTES (16*1024 + 128*1024 + 32*1024 + 256*4 + 16*4)

__global__ __launch_bounds__(THREADS, 1)
void Attention_23991687315844_kernel(const float* __restrict__ q,
                                     const float* __restrict__ k,
                                     const float* __restrict__ v,
                                     const unsigned char* __restrict__ mask,
                                     float* __restrict__ out,
                                     float* __restrict__ attn)
{
    bf16_t* sQ   = reinterpret_cast<bf16_t*>(smem);
    float*  sS   = reinterpret_cast<float*>(smem + 16 * 1024);
    bf16_t* sVt  = reinterpret_cast<bf16_t*>(smem + 16 * 1024 + 128 * 1024);
    float*  sRed = reinterpret_cast<float*>(smem + 176 * 1024);
    float*  sInv = sRed + 256;

    const int tid  = threadIdx.x;
    const int wave = tid >> 5;
    const int lane = tid & 31;
    const int lrow = lane & 15;   // M for A operand / N for B operand / N for C,D
    const int lgrp = lane >> 4;   // K-half selector for A/B, row-half for C/D

    const int b  = blockIdx.x / KTILES;
    const int qt = blockIdx.x % KTILES;
    const int q0 = qt * QT;

    const float* qb = q + (size_t)b * S_ * D_;
    const float* kb = k + (size_t)b * S_ * D_;
    const float* vb = v + (size_t)b * S_ * D_;
    const unsigned char* mb = mask + (size_t)b * S_ * S_;
    float* outb  = out  + (size_t)b * S_ * D_;
    float* attnb = attn + (size_t)b * S_ * S_;

    // ---------------- Phase 0: stage Q tile (f32 -> bf16) ----------------
    for (int idx = tid * 4; idx < QT * D_; idx += THREADS * 4) {
        const float4 f = *reinterpret_cast<const float4*>(qb + (size_t)q0 * D_ + idx);
        sQ[idx + 0] = (bf16_t)f.x;
        sQ[idx + 1] = (bf16_t)f.y;
        sQ[idx + 2] = (bf16_t)f.z;
        sQ[idx + 3] = (bf16_t)f.w;
    }
    __syncthreads();

    // ---------------- Phase 1: S = scale * (Q @ K^T), masked ----------------
    const float scale = 0.04419417382415922f; // 1/sqrt(512)
    for (int kt = wave; kt < KTILES; kt += NWAVES) {
        v8f c = {};
        const bf16_t* qrow = sQ + lrow * D_;                         // A: M = lrow
        const float*  krow = kb + (size_t)(kt * 16 + lrow) * D_;     // B: N = lrow (K rows)
        #pragma unroll
        for (int ch = 0; ch < QK_CHUNKS; ++ch) {
            v16bf a  = ld16_bf16(qrow + ch * 32, lgrp);
            v16bf bm = cvt16_f32(krow + ch * 32, lgrp);
            c = __builtin_amdgcn_wmma_f32_16x16x32_bf16(
                    false, a, false, bm, (short)0, c, false, false);
        }
        const int j = kt * 16 + lrow;   // score column
        #pragma unroll
        for (int r = 0; r < 8; ++r) {
            const int m = r + 8 * lgrp; // score row within tile
            float val = c[r] * scale;
            if (mb[(size_t)(q0 + m) * S_ + j]) val = -INFINITY;
            sS[m * S_ + j] = val;
        }
    }
    __syncthreads();

    // ---------------- Phase 2: row softmax over 2048 columns ----------------
    const int srow = tid >> 4;   // 16 threads per row
    const int ssub = tid & 15;
    float pmax = -INFINITY;
    for (int j = ssub; j < S_; j += 16) pmax = fmaxf(pmax, sS[srow * S_ + j]);
    sRed[tid] = pmax;
    __syncthreads();
    float rmax = -INFINITY;
    #pragma unroll
    for (int i = 0; i < 16; ++i) rmax = fmaxf(rmax, sRed[srow * 16 + i]);
    __syncthreads();                     // sRed reuse
    float psum = 0.f;
    for (int j = ssub; j < S_; j += 16) {
        const float p = __expf(sS[srow * S_ + j] - rmax);
        sS[srow * S_ + j] = p;           // keep unnormalized exp for P@V
        psum += p;
    }
    sRed[tid] = psum;
    __syncthreads();
    float rsum = 0.f;
    #pragma unroll
    for (int i = 0; i < 16; ++i) rsum += sRed[srow * 16 + i];
    if (ssub == 0) sInv[srow] = 1.0f / rsum;
    __syncthreads();

    // Write normalized attention probabilities (coalesced float4).
    for (int idx = tid * 4; idx < QT * S_; idx += THREADS * 4) {
        const int r = idx >> 11;         // idx / 2048
        const float inv = sInv[r];
        const float4 p = *reinterpret_cast<const float4*>(sS + idx);
        float4 o;
        o.x = p.x * inv; o.y = p.y * inv; o.z = p.z * inv; o.w = p.w * inv;
        *reinterpret_cast<float4*>(attnb + (size_t)q0 * S_ + idx) = o;
    }

    // ---------------- Phase 3: O = P @ V (wave owns 64 output columns) ----------------
    v8f acc0 = {}, acc1 = {}, acc2 = {}, acc3 = {};
    const int n0 = wave * 64;
    for (int kc = 0; kc < PV_CHUNKS; ++kc) {
        __syncthreads();  // previous chunk's sVt reads complete
        // Stage V[kc*32 .. kc*32+31][0..511] transposed into sVt[d][kk] (bf16)
        for (int idx = tid * 4; idx < 32 * D_; idx += THREADS * 4) {
            const int kk = idx / D_;
            const int d0 = idx % D_;
            const float4 f = *reinterpret_cast<const float4*>(
                vb + (size_t)(kc * 32 + kk) * D_ + d0);
            sVt[(d0 + 0) * 32 + kk] = (bf16_t)f.x;
            sVt[(d0 + 1) * 32 + kk] = (bf16_t)f.y;
            sVt[(d0 + 2) * 32 + kk] = (bf16_t)f.z;
            sVt[(d0 + 3) * 32 + kk] = (bf16_t)f.w;
        }
        __syncthreads();

        // A operand: P row (f32 exp values in LDS) -> bf16
        const v16bf a = cvt16_f32(sS + lrow * S_ + kc * 32, lgrp);
        // B operands: transposed V columns, K = g*8+0..7 and 16+g*8+0..7
        const bf16_t* vc = sVt + (size_t)(n0 + lrow) * 32;
        const v16bf b0 = ld16_bf16(vc, lgrp);
        const v16bf b1 = ld16_bf16(vc + 16 * 32, lgrp);
        const v16bf b2 = ld16_bf16(vc + 32 * 32, lgrp);
        const v16bf b3 = ld16_bf16(vc + 48 * 32, lgrp);

        acc0 = __builtin_amdgcn_wmma_f32_16x16x32_bf16(false, a, false, b0, (short)0, acc0, false, false);
        acc1 = __builtin_amdgcn_wmma_f32_16x16x32_bf16(false, a, false, b1, (short)0, acc1, false, false);
        acc2 = __builtin_amdgcn_wmma_f32_16x16x32_bf16(false, a, false, b2, (short)0, acc2, false, false);
        acc3 = __builtin_amdgcn_wmma_f32_16x16x32_bf16(false, a, false, b3, (short)0, acc3, false, false);
    }

    // Fold 1/rowsum into O and store.
    #pragma unroll
    for (int r = 0; r < 8; ++r) {
        const int m = r + 8 * lgrp;
        const float inv = sInv[m];
        float* orow = outb + (size_t)(q0 + m) * D_ + n0 + lrow;
        orow[0]  = acc0[r] * inv;
        orow[16] = acc1[r] * inv;
        orow[32] = acc2[r] * inv;
        orow[48] = acc3[r] * inv;
    }
}

extern "C" void kernel_launch(void* const* d_in, const int* in_sizes, int n_in,
                              void* d_out, int out_size, void* d_ws, size_t ws_size,
                              hipStream_t stream) {
    (void)in_sizes; (void)n_in; (void)out_size; (void)d_ws; (void)ws_size;
    const float* q = (const float*)d_in[0];
    const float* k = (const float*)d_in[1];
    const float* v = (const float*)d_in[2];
    const unsigned char* mask = (const unsigned char*)d_in[3];

    float* out  = (float*)d_out;                       // (B,S,D)
    float* attn = out + (size_t)B_ * S_ * D_;          // (B,S,S), concatenated

    hipFuncSetAttribute((const void*)Attention_23991687315844_kernel,
                        hipFuncAttributeMaxDynamicSharedMemorySize, SMEM_BYTES);

    dim3 grid(B_ * KTILES);   // 2048 workgroups: one 16-row Q tile each
    dim3 block(THREADS);      // 8 waves (wave32)
    Attention_23991687315844_kernel<<<grid, block, SMEM_BYTES, stream>>>(
        q, k, v, mask, out, attn);
}
